// FSSLoss_47132971106458
// MI455X (gfx1250) — matrix operand order repaired and checked by reading, not compiled, not run on hardware
//
#include <hip/hip_runtime.h>

typedef __attribute__((ext_vector_type(16))) _Float16 v16h;
typedef __attribute__((ext_vector_type(8)))  float    v8f;

#define H_IMG 1024
#define W_IMG 1024
#define SHARP 10.0f
#define N_SLOTS 288   // 3 thr * 16 img * 3 windows * {mse,ref}

__global__ void fss_zero_ws(float* ws) {
  int i = blockIdx.x * blockDim.x + threadIdx.x;
  if (i < N_SLOTS) ws[i] = 0.0f;
}

__device__ __forceinline__ float sigmoid10(float x, float thr) {
#if __has_builtin(__builtin_amdgcn_tanhf)
  // sigma(k*z) = 0.5 + 0.5*tanh(k*z/2): single v_tanh_f32 (gfx1250 trans op)
  return fmaf(0.5f, __builtin_amdgcn_tanhf((0.5f * SHARP) * (x - thr)), 0.5f);
#else
  float e = __expf(-SHARP * (x - thr));
  return __builtin_amdgcn_rcpf(1.0f + e);
#endif
}

// grid: (16 images, 16 col-strips-of-64, 3 thresholds); block: 128 (4 waves)
// Per chunk of 16 rows: block stages sigmoid/threshold f16 fields for its
// 96-column span in LDS (double-buffered), waves build WMMA B tiles from LDS,
// banded A matrices compute horizontal box sums (k=9,33), vertical box sums
// via per-wave LDS rings with O(1) running-sum updates; k=1 during staging.
__global__ __launch_bounds__(128)
void fss_main(const float* __restrict__ fcst,
              const float* __restrict__ obs,
              float* __restrict__ ws) {
  __shared__ float ring9f [4][64][16];
  __shared__ float ring9o [4][64][16];
  __shared__ float ring33f[4][64][16];
  __shared__ float ring33o[4][64][16];               // 64 KB
  __shared__ __align__(32) _Float16 sfs[2][16][96];  // staged f (double buffer)
  __shared__ __align__(32) _Float16 sos[2][16][96];  // staged o            12 KB

  const int n    = blockIdx.x;
  const int t    = blockIdx.z;
  const float thr = 0.5f * (float)t;            // 0.0, 0.5, 1.0
  const int tid  = threadIdx.x;
  const int w    = tid >> 5;
  const int lane = tid & 31;
  const int hh   = lane >> 4;
  const int lm   = lane & 15;
  const int blockc0 = blockIdx.y * 64;          // block owns cols [blockc0, blockc0+64)
  const bool bedge  = (blockIdx.y == 0) || (blockIdx.y == 15);

  const float* __restrict__ fbase = fcst + ((size_t)n << 20);
  const float* __restrict__ obase = obs  + ((size_t)n << 20);

  // ---- constant banded A matrices (16x32 f16, A-layout per lane) ----
  v16h a9l, a9r, a33l, a33r;
  for (int j = 0; j < 16; ++j) {
    int K = 16 * hh + j;
    int m = lm;
    a33l[j] = (_Float16)((K >= m) ? 1.0f : 0.0f);
    a33r[j] = (_Float16)((K <= m) ? 1.0f : 0.0f);
    a9l[j]  = (_Float16)((K >= m + 12 && K <= m + 20) ? 1.0f : 0.0f);
    a9r[j]  = (_Float16)((K >= m - 20 && K <= m - 12) ? 1.0f : 0.0f);
  }

  // unified consumer: group 0 (lanes 0-15) -> k9, group 1 -> k33
  float* __restrict__ rF = hh ? &ring33f[w][0][0] : &ring9f[w][0][0];
  float* __restrict__ rO = hh ? &ring33o[w][0][0] : &ring9o[w][0][0];
  const int   dp   = hh ? 16 : 4;
  const int   dm   = hh ? 17 : 5;
  const float invA = hh ? (1.0f / 1089.0f) : (1.0f / 81.0f);

  float vF = 0.f, vO = 0.f;        // vertical running sums
  float sM = 0.f, sR = 0.f;        // k9 (half0) / k33 (half1) stats
  float s1m = 0.f, s1r = 0.f;      // k=1 stats (per thread, from staging)

  // staged index 0 == abs col (blockc0 - 16); wave's tiles:
  const int s0L = (w << 4) + (hh << 4);   // left tile span (always used)
  const int s0R = s0L + 32;               // right tile lower K-half (hh==0 only)
  const int m0  = hh << 3;

  for (int c = 0; c <= 64; ++c) {
    const int r0 = c << 4;
    const int p  = c & 1;

    if (c < 64) {
      // ---- stage f/o (f16) for block cols [blockc0-16, blockc0+80), 16 rows ----
      for (int it = 0; it < 3; ++it) {
        int idx  = tid + (it << 7);          // 0..383 == srow*24 + c4
        int srow = idx / 24;
        int c4   = idx - srow * 24;          // float4 group within 96-col span
        int col  = blockc0 - 16 + (c4 << 2);
        const float* frowp = fbase + ((size_t)(r0 + srow) << 10);
        const float* orowp = obase + ((size_t)(r0 + srow) << 10);
        float fa[4], ba[4], mk[4];
        if (!bedge) {
          float4 a = *(const float4*)(frowp + col);
          float4 b = *(const float4*)(orowp + col);
          fa[0] = a.x; fa[1] = a.y; fa[2] = a.z; fa[3] = a.w;
          ba[0] = b.x; ba[1] = b.y; ba[2] = b.z; ba[3] = b.w;
          mk[0] = mk[1] = mk[2] = mk[3] = 1.0f;
          if (c < 62 && it == 0) {
            __builtin_prefetch(frowp + (16 << 10) + col, 0, 1);
            __builtin_prefetch(orowp + (16 << 10) + col, 0, 1);
          }
        } else {
          for (int u = 0; u < 4; ++u) {
            int cc = col + u;
            bool ib = (cc >= 0) && (cc < W_IMG);
            mk[u] = ib ? 1.0f : 0.0f;
            fa[u] = ib ? frowp[cc] : 0.0f;
            ba[u] = ib ? orowp[cc] : 0.0f;
          }
        }
        float selOwn = (c4 >= 4 && c4 < 20) ? 1.0f : 0.0f;  // owned columns
        for (int u = 0; u < 4; ++u) {
          float f = mk[u] * sigmoid10(fa[u], thr);
          float o = (ba[u] > thr) ? mk[u] : 0.0f;
          sfs[p][srow][(c4 << 2) + u] = (_Float16)f;
          sos[p][srow][(c4 << 2) + u] = (_Float16)o;
          float d = f - o;
          s1m += selOwn * d * d;
          s1r += selOwn * (f * f + o * o);
        }
      }
      __syncthreads();   // staging visible to all waves (double buffer => 1 barrier)

      // ---- B tiles from staged LDS (v16h = 2x ds_load_b128) ----
      v16h bLf = *(const v16h*)&sfs[p][lm][s0L];
      v16h bLo = *(const v16h*)&sos[p][lm][s0L];
      v16h bRf = {}, bRo = {};
      if (hh == 0) {     // right-tile upper K-half has all-zero A bands: skip
        bRf = *(const v16h*)&sfs[p][lm][s0R];
        bRo = *(const v16h*)&sos[p][lm][s0R];
      }

      // ---- horizontal box sums via WMMA; store each pair as it retires ----
      const unsigned rr = (unsigned)(r0 + lm) & 63u;
      {
        v8f h9f = {}, h9o = {};
        h9f = __builtin_amdgcn_wmma_f32_16x16x32_f16(false, a9l, false, bLf, (short)0, h9f, false, false);
        h9f = __builtin_amdgcn_wmma_f32_16x16x32_f16(false, a9r, false, bRf, (short)0, h9f, false, false);
        h9o = __builtin_amdgcn_wmma_f32_16x16x32_f16(false, a9l, false, bLo, (short)0, h9o, false, false);
        h9o = __builtin_amdgcn_wmma_f32_16x16x32_f16(false, a9r, false, bRo, (short)0, h9o, false, false);
        for (int j = 0; j < 8; ++j) {
          ring9f[w][rr][m0 + j] = h9f[j];
          ring9o[w][rr][m0 + j] = h9o[j];
        }
      }
      {
        v8f h33f = {}, h33o = {};
        h33f = __builtin_amdgcn_wmma_f32_16x16x32_f16(false, a33l, false, bLf, (short)0, h33f, false, false);
        h33f = __builtin_amdgcn_wmma_f32_16x16x32_f16(false, a33r, false, bRf, (short)0, h33f, false, false);
        h33o = __builtin_amdgcn_wmma_f32_16x16x32_f16(false, a33l, false, bLo, (short)0, h33o, false, false);
        h33o = __builtin_amdgcn_wmma_f32_16x16x32_f16(false, a33r, false, bRo, (short)0, h33o, false, false);
        for (int j = 0; j < 8; ++j) {
          ring33f[w][rr][m0 + j] = h33f[j];
          ring33o[w][rr][m0 + j] = h33o[j];
        }
      }
    } else {
      // c == 64: flush zero rows (rows 1024..1039) so tail windows close
      const unsigned rr = (unsigned)(r0 + lm) & 63u;
      for (int j = 0; j < 8; ++j) {
        ring9f [w][rr][m0 + j] = 0.0f;
        ring9o [w][rr][m0 + j] = 0.0f;
        ring33f[w][rr][m0 + j] = 0.0f;
        ring33o[w][rr][m0 + j] = 0.0f;
      }
    }
    // per-wave private rings; same-wave LDS ops are in-order -> no extra barrier

    // ---- vertical running box sums; emit output rows [r0-16, r0) ----
    if (c < 3) {
      for (int i = 0; i < 16; ++i) {
        int q = r0 - 16 + i;
        int rp = q + dp, rm = q - dm;
        float gp = (rp >= 0) ? 1.0f : 0.0f;
        float gm = (rm >= 0) ? 1.0f : 0.0f;
        int rpc = (rp < 0 ? 0 : rp) & 63;
        int rmc = (rm < 0 ? 0 : rm) & 63;
        vF += gp * rF[rpc * 16 + lm] - gm * rF[rmc * 16 + lm];
        vO += gp * rO[rpc * 16 + lm] - gm * rO[rmc * 16 + lm];
        if (q >= 0) {
          float F = vF * invA, O = vO * invA, d = F - O;
          sM += d * d;
          sR += F * F + O * O;
        }
      }
    } else {
      for (int i = 0; i < 16; ++i) {
        int q = r0 - 16 + i;       // q >= 32: all indices in range
        vF += rF[((q + dp) & 63) * 16 + lm] - rF[((q - dm) & 63) * 16 + lm];
        vO += rO[((q + dp) & 63) * 16 + lm] - rO[((q - dm) & 63) * 16 + lm];
        float F = vF * invA, O = vO * invA, d = F - O;
        sM += d * d;
        sR += F * F + O * O;
      }
    }
  }

  // ---- reductions: s1 across full wave; sM/sR within 16-lane halves ----
  for (int off = 16; off > 0; off >>= 1) {
    s1m += __shfl_xor(s1m, off, 32);
    s1r += __shfl_xor(s1r, off, 32);
  }
  for (int off = 8; off > 0; off >>= 1) {
    sM += __shfl_xor(sM, off, 32);
    sR += __shfl_xor(sR, off, 32);
  }
  const int base = ((t * 16 + n) * 3) * 2;  // k-index {0:k1, 1:k9, 2:k33} x {mse,ref}
  if (lane == 0) {
    atomicAdd(&ws[base + 0], s1m);
    atomicAdd(&ws[base + 1], s1r);
    atomicAdd(&ws[base + 2], sM);   // k9 (half 0)
    atomicAdd(&ws[base + 3], sR);
  }
  if (lane == 16) {
    atomicAdd(&ws[base + 4], sM);   // k33 (half 1)
    atomicAdd(&ws[base + 5], sR);
  }
}

__global__ void fss_finalize(const float* __restrict__ ws, float* __restrict__ out) {
  __shared__ float acc[144];
  int i = threadIdx.x;
  if (i < 144) {
    float sm = ws[2 * i + 0];
    float sr = ws[2 * i + 1];
    // fss = 1 - (Smse/HW) / (Sref/HW + eps) = 1 - Smse / (Sref + eps*HW)
    acc[i] = 1.0f - sm / (sr + 1e-8f * 1048576.0f);
  }
  __syncthreads();
  if (i == 0) {
    float s = 0.0f;
    for (int j = 0; j < 144; ++j) s += acc[j];
    out[0] = 1.0f - s / 144.0f;
  }
}

extern "C" void kernel_launch(void* const* d_in, const int* in_sizes, int n_in,
                              void* d_out, int out_size, void* d_ws, size_t ws_size,
                              hipStream_t stream) {
  (void)in_sizes; (void)n_in; (void)out_size; (void)ws_size;
  const float* fcst = (const float*)d_in[0];
  const float* obs  = (const float*)d_in[1];
  float* out = (float*)d_out;
  float* ws  = (float*)d_ws;

  fss_zero_ws<<<1, N_SLOTS, 0, stream>>>(ws);
  dim3 grid(16, 16, 3);  // images x col-strips(64 cols) x thresholds
  fss_main<<<grid, 128, 0, stream>>>(fcst, obs, ws);
  fss_finalize<<<1, 256, 0, stream>>>(ws, out);
}
